// ALiBiAttention_44092134260809
// MI455X (gfx1250) — matrix operand order repaired and checked by reading, not compiled, not run on hardware
//
#include <hip/hip_runtime.h>
#include <hip/hip_fp16.h>

typedef __attribute__((ext_vector_type(16))) _Float16 v16h;
typedef __attribute__((ext_vector_type(4)))  _Float16 v4h;
typedef __attribute__((ext_vector_type(8)))  float    v8f;
typedef __attribute__((ext_vector_type(4)))  float    v4f;

#define DIMC   1024
#define NSEQ   2048
#define NHEADS 16
#define HDIM   64
#define BATCH  2
#define MROWS  (BATCH * NSEQ)   // 4096 rows across batch

// ---------------------------------------------------------------------------
// Kernel 0: one-shot f32 -> f16 conversion (vectorized 4-wide).
// ---------------------------------------------------------------------------
__global__ __launch_bounds__(256)
void cvt_f16_kernel(const float* __restrict__ src, _Float16* __restrict__ dst, int n4)
{
    const int i = blockIdx.x * blockDim.x + threadIdx.x;
    if (i < n4) {
        const v4f s = ((const v4f*)src)[i];
        v4h d;
        d[0] = (_Float16)s[0]; d[1] = (_Float16)s[1];
        d[2] = (_Float16)s[2]; d[3] = (_Float16)s[3];
        ((v4h*)dst)[i] = d;
    }
}

// ---------------------------------------------------------------------------
// Kernel 1: fused QKV projection, pure f16 operands.  y = x @ W^T.
// One wave computes a 32x64 output tile (2 M-frags x 4 N-frags, 8 WMMAs/K-step).
// Q/K stored [B,H,N,D]; V stored transposed [B,H,D,N] so the attention
// kernel's PV B-fragments are contiguous.
// ---------------------------------------------------------------------------
__global__ __launch_bounds__(32)
void qkv_proj_kernel(const _Float16* __restrict__ xh,
                     const _Float16* __restrict__ Wqh,
                     const _Float16* __restrict__ Wkh,
                     const _Float16* __restrict__ Wvh,
                     _Float16* __restrict__ Qb,
                     _Float16* __restrict__ Kb,
                     _Float16* __restrict__ Vt)
{
    const int lane = threadIdx.x & 31;
    const int lm   = lane & 15;
    const int hi   = lane >> 4;
    const int n0   = blockIdx.x * 64;
    const int m0   = blockIdx.y * 32;
    const int which = blockIdx.z;

    const _Float16* W = (which == 0) ? Wqh : (which == 1) ? Wkh : Wvh;

    const _Float16* arow0 = xh + (size_t)(m0 + lm) * DIMC;
    const _Float16* arow1 = xh + (size_t)(m0 + 16 + lm) * DIMC;
    const _Float16* wrow0 = W + (size_t)(n0 +  0 + lm) * DIMC;
    const _Float16* wrow1 = W + (size_t)(n0 + 16 + lm) * DIMC;
    const _Float16* wrow2 = W + (size_t)(n0 + 32 + lm) * DIMC;
    const _Float16* wrow3 = W + (size_t)(n0 + 48 + lm) * DIMC;

    v8f acc[2][4] = {};
    for (int kk = 0; kk < DIMC; kk += 32) {
        v16h a0, a1, b0, b1, b2, b3;
#pragma unroll
        for (int h = 0; h < 16; ++h) {
            const int ka = kk + (hi << 4) + h;                        // A layout
            a0[h] = arow0[ka];
            a1[h] = arow1[ka];
            const int kb = kk + (hi << 4) + h;                        // B layout
            b0[h] = wrow0[kb];
            b1[h] = wrow1[kb];
            b2[h] = wrow2[kb];
            b3[h] = wrow3[kb];
        }
        acc[0][0] = __builtin_amdgcn_wmma_f32_16x16x32_f16(false, a0, false, b0, (short)0, acc[0][0], false, false);
        acc[0][1] = __builtin_amdgcn_wmma_f32_16x16x32_f16(false, a0, false, b1, (short)0, acc[0][1], false, false);
        acc[0][2] = __builtin_amdgcn_wmma_f32_16x16x32_f16(false, a0, false, b2, (short)0, acc[0][2], false, false);
        acc[0][3] = __builtin_amdgcn_wmma_f32_16x16x32_f16(false, a0, false, b3, (short)0, acc[0][3], false, false);
        acc[1][0] = __builtin_amdgcn_wmma_f32_16x16x32_f16(false, a1, false, b0, (short)0, acc[1][0], false, false);
        acc[1][1] = __builtin_amdgcn_wmma_f32_16x16x32_f16(false, a1, false, b1, (short)0, acc[1][1], false, false);
        acc[1][2] = __builtin_amdgcn_wmma_f32_16x16x32_f16(false, a1, false, b2, (short)0, acc[1][2], false, false);
        acc[1][3] = __builtin_amdgcn_wmma_f32_16x16x32_f16(false, a1, false, b3, (short)0, acc[1][3], false, false);
    }

    // C layout per frag: lane holds col (t*16+lm), rows (mi*16 + r + 8*hi).
    if (which < 2) {
        _Float16* Out = (which == 0) ? Qb : Kb;
#pragma unroll
        for (int mi = 0; mi < 2; ++mi)
#pragma unroll
            for (int t = 0; t < 4; ++t)
#pragma unroll
                for (int r = 0; r < 8; ++r) {
                    const int mg   = m0 + mi * 16 + r + 8 * hi;
                    const int ng   = n0 + t * 16 + lm;
                    const int bI   = mg >> 11;
                    const int seq  = mg & (NSEQ - 1);
                    const int head = ng >> 6;
                    const int d    = ng & (HDIM - 1);
                    Out[(((size_t)(bI * NHEADS + head) * NSEQ) + seq) * HDIM + d] =
                        (_Float16)acc[mi][t][r];
                }
    } else {
        // V transposed: [B,H,D,N]; r-loop writes contiguous seq positions.
#pragma unroll
        for (int mi = 0; mi < 2; ++mi)
#pragma unroll
            for (int t = 0; t < 4; ++t)
#pragma unroll
                for (int r = 0; r < 8; ++r) {
                    const int mg   = m0 + mi * 16 + r + 8 * hi;
                    const int ng   = n0 + t * 16 + lm;
                    const int bI   = mg >> 11;
                    const int seq  = mg & (NSEQ - 1);
                    const int head = ng >> 6;
                    const int d    = ng & (HDIM - 1);
                    Vt[(((size_t)(bI * NHEADS + head) * HDIM) + d) * NSEQ + seq] =
                        (_Float16)acc[mi][t][r];
                }
    }
}

// ---------------------------------------------------------------------------
// Kernel 2: flash-style ALiBi attention.  One wave per (b, head, 16-query tile).
// score = (q.k)*8 + slope*(j-i); online softmax; O = P @ V.
// ---------------------------------------------------------------------------
__global__ __launch_bounds__(32)
void alibi_attn_kernel(const _Float16* __restrict__ Qb,
                       const _Float16* __restrict__ Kb,
                       const _Float16* __restrict__ Vt,
                       _Float16* __restrict__ Ob)
{
    __shared__ _Float16 ldsP[16][32];

    const int lane = threadIdx.x & 31;
    const int lm   = lane & 15;
    const int hi   = lane >> 4;
    const int qt   = blockIdx.x;
    const int head = blockIdx.y;
    const int bI   = blockIdx.z;

    const float  slope = exp2f(-0.5f * (float)(head + 1));
    const size_t base  = (size_t)(bI * NHEADS + head) * NSEQ * HDIM; // same size for V^T
    const int    i0    = qt * 16;

    const _Float16* qrow = Qb + base + (size_t)(i0 + lm) * HDIM;
    v16h aq0, aq1;
#pragma unroll
    for (int h = 0; h < 16; ++h) {
        const int k = (hi << 4) + h;
        aq0[h] = qrow[k];
        aq1[h] = qrow[32 + k];
    }

    float m_i[8], l_i[8];
    v8f oacc[4] = {};
#pragma unroll
    for (int r = 0; r < 8; ++r) { m_i[r] = -1e30f; l_i[r] = 0.0f; }

    const int myrow = i0 + 8 * hi;

    for (int j0 = 0; j0 < NSEQ; j0 += 32) {
        float s0[8], s1[8];
#pragma unroll
        for (int sub = 0; sub < 2; ++sub) {
            const _Float16* krow = Kb + base + (size_t)(j0 + sub * 16 + lm) * HDIM;
            v16h bk0, bk1;
#pragma unroll
            for (int h = 0; h < 16; ++h) {
                const int k = (hi << 4) + h;
                bk0[h] = krow[k];
                bk1[h] = krow[32 + k];
            }
            v8f sacc = {};
            sacc = __builtin_amdgcn_wmma_f32_16x16x32_f16(false, aq0, false, bk0, (short)0, sacc, false, false);
            sacc = __builtin_amdgcn_wmma_f32_16x16x32_f16(false, aq1, false, bk1, (short)0, sacc, false, false);
            float*    s = sub ? s1 : s0;
            const int j = j0 + sub * 16 + lm;
#pragma unroll
            for (int r = 0; r < 8; ++r) {
                const int i = myrow + r;
                s[r] = sacc[r] * 8.0f + slope * (float)(j - i);
            }
        }

        float alpha[8];
#pragma unroll
        for (int r = 0; r < 8; ++r) {
            float v = fmaxf(s0[r], s1[r]);
#pragma unroll
            for (int mask = 1; mask < 16; mask <<= 1)
                v = fmaxf(v, __shfl_xor(v, mask, 32));
            const float m_new = fmaxf(m_i[r], v);
            alpha[r] = __expf(m_i[r] - m_new);
            const float p0 = __expf(s0[r] - m_new);
            const float p1 = __expf(s1[r] - m_new);
            float psum = p0 + p1;
#pragma unroll
            for (int mask = 1; mask < 16; mask <<= 1)
                psum += __shfl_xor(psum, mask, 32);
            l_i[r] = l_i[r] * alpha[r] + psum;
            m_i[r] = m_new;
            ldsP[r + 8 * hi][lm]      = (_Float16)p0;
            ldsP[r + 8 * hi][16 + lm] = (_Float16)p1;
        }
#pragma unroll
        for (int dc = 0; dc < 4; ++dc)
#pragma unroll
            for (int r = 0; r < 8; ++r)
                oacc[dc][r] *= alpha[r];

        __syncthreads();
        v16h ap;
#pragma unroll
        for (int h = 0; h < 16; ++h) {
            const int slot = (hi << 4) + h;
            ap[h] = ldsP[lm][slot];
        }
        __syncthreads();

        // PV with transposed V: contiguous 16-half runs per fragment.
#pragma unroll
        for (int dc = 0; dc < 4; ++dc) {
            const _Float16* vrow = Vt + base + (size_t)(dc * 16 + lm) * NSEQ + j0 + (hi << 4);
            v16h bv;
#pragma unroll
            for (int h = 0; h < 16; ++h)
                bv[h] = vrow[h];
            oacc[dc] = __builtin_amdgcn_wmma_f32_16x16x32_f16(false, ap, false, bv,
                                                              (short)0, oacc[dc], false, false);
        }
    }

#pragma unroll
    for (int r = 0; r < 8; ++r) {
        const float inv = 1.0f / l_i[r];
        const int   seq = myrow + r;
        const size_t ob = ((size_t)bI * NSEQ + seq) * DIMC + head * HDIM + lm;
        Ob[ob +  0] = (_Float16)(oacc[0][r] * inv);
        Ob[ob + 16] = (_Float16)(oacc[1][r] * inv);
        Ob[ob + 32] = (_Float16)(oacc[2][r] * inv);
        Ob[ob + 48] = (_Float16)(oacc[3][r] * inv);
    }
}

// ---------------------------------------------------------------------------
// Kernel 3: output projection.  out = O @ Wp^T + bp, fp32 result.
// 32x64 tile per wave, f16 operands.
// ---------------------------------------------------------------------------
__global__ __launch_bounds__(32)
void out_proj_kernel(const _Float16* __restrict__ Ob,
                     const _Float16* __restrict__ Wph,
                     const float* __restrict__ bp,
                     float* __restrict__ out)
{
    const int lane = threadIdx.x & 31;
    const int lm   = lane & 15;
    const int hi   = lane >> 4;
    const int n0   = blockIdx.x * 64;
    const int m0   = blockIdx.y * 32;

    const _Float16* arow0 = Ob + (size_t)(m0 + lm) * DIMC;
    const _Float16* arow1 = Ob + (size_t)(m0 + 16 + lm) * DIMC;
    const _Float16* wrow0 = Wph + (size_t)(n0 +  0 + lm) * DIMC;
    const _Float16* wrow1 = Wph + (size_t)(n0 + 16 + lm) * DIMC;
    const _Float16* wrow2 = Wph + (size_t)(n0 + 32 + lm) * DIMC;
    const _Float16* wrow3 = Wph + (size_t)(n0 + 48 + lm) * DIMC;

    v8f acc[2][4] = {};
    for (int kk = 0; kk < DIMC; kk += 32) {
        v16h a0, a1, b0, b1, b2, b3;
#pragma unroll
        for (int h = 0; h < 16; ++h) {
            const int ka = kk + (hi << 4) + h;
            a0[h] = arow0[ka];
            a1[h] = arow1[ka];
            const int kb = kk + (hi << 4) + h;
            b0[h] = wrow0[kb];
            b1[h] = wrow1[kb];
            b2[h] = wrow2[kb];
            b3[h] = wrow3[kb];
        }
        acc[0][0] = __builtin_amdgcn_wmma_f32_16x16x32_f16(false, a0, false, b0, (short)0, acc[0][0], false, false);
        acc[0][1] = __builtin_amdgcn_wmma_f32_16x16x32_f16(false, a0, false, b1, (short)0, acc[0][1], false, false);
        acc[0][2] = __builtin_amdgcn_wmma_f32_16x16x32_f16(false, a0, false, b2, (short)0, acc[0][2], false, false);
        acc[0][3] = __builtin_amdgcn_wmma_f32_16x16x32_f16(false, a0, false, b3, (short)0, acc[0][3], false, false);
        acc[1][0] = __builtin_amdgcn_wmma_f32_16x16x32_f16(false, a1, false, b0, (short)0, acc[1][0], false, false);
        acc[1][1] = __builtin_amdgcn_wmma_f32_16x16x32_f16(false, a1, false, b1, (short)0, acc[1][1], false, false);
        acc[1][2] = __builtin_amdgcn_wmma_f32_16x16x32_f16(false, a1, false, b2, (short)0, acc[1][2], false, false);
        acc[1][3] = __builtin_amdgcn_wmma_f32_16x16x32_f16(false, a1, false, b3, (short)0, acc[1][3], false, false);
    }

#pragma unroll
    for (int t = 0; t < 4; ++t) {
        const float bias = bp[n0 + t * 16 + lm];
#pragma unroll
        for (int mi = 0; mi < 2; ++mi)
#pragma unroll
            for (int r = 0; r < 8; ++r) {
                const int mg = m0 + mi * 16 + r + 8 * hi;
                out[(size_t)mg * DIMC + n0 + t * 16 + lm] = acc[mi][t][r] + bias;
            }
    }
}

// ---------------------------------------------------------------------------
extern "C" void kernel_launch(void* const* d_in, const int* in_sizes, int n_in,
                              void* d_out, int out_size, void* d_ws, size_t ws_size,
                              hipStream_t stream)
{
    const float* x  = (const float*)d_in[0];
    const float* Wq = (const float*)d_in[1];
    const float* Wk = (const float*)d_in[2];
    const float* Wv = (const float*)d_in[3];
    const float* Wp = (const float*)d_in[4];
    const float* bp = (const float*)d_in[5];

    const size_t MB = 1024 * 1024;
    char* ws = (char*)d_ws;
    _Float16* Qb  = (_Float16*)(ws +  0 * MB);   // 8 MiB  [B,H,N,D]
    _Float16* Kb  = (_Float16*)(ws +  8 * MB);   // 8 MiB  [B,H,N,D]
    _Float16* Vt  = (_Float16*)(ws + 16 * MB);   // 8 MiB  [B,H,D,N]
    _Float16* Ob  = (_Float16*)(ws + 24 * MB);   // 8 MiB  [B,N,C]
    _Float16* xh  = (_Float16*)(ws + 32 * MB);   // 8 MiB
    _Float16* Wqh = (_Float16*)(ws + 40 * MB);   // 2 MiB each
    _Float16* Wkh = (_Float16*)(ws + 42 * MB);
    _Float16* Wvh = (_Float16*)(ws + 44 * MB);
    _Float16* Wph = (_Float16*)(ws + 46 * MB);

    const int nX4 = MROWS * DIMC / 4;   // 1,048,576
    const int nW4 = DIMC * DIMC / 4;    //   262,144
    cvt_f16_kernel<<<dim3((nX4 + 255) / 256), dim3(256), 0, stream>>>(x,  xh,  nX4);
    cvt_f16_kernel<<<dim3((nW4 + 255) / 256), dim3(256), 0, stream>>>(Wq, Wqh, nW4);
    cvt_f16_kernel<<<dim3((nW4 + 255) / 256), dim3(256), 0, stream>>>(Wk, Wkh, nW4);
    cvt_f16_kernel<<<dim3((nW4 + 255) / 256), dim3(256), 0, stream>>>(Wv, Wvh, nW4);
    cvt_f16_kernel<<<dim3((nW4 + 255) / 256), dim3(256), 0, stream>>>(Wp, Wph, nW4);

    const dim3 blk(32);
    qkv_proj_kernel<<<dim3(DIMC / 64, MROWS / 32, 3), blk, 0, stream>>>(
        xh, Wqh, Wkh, Wvh, Qb, Kb, Vt);
    alibi_attn_kernel<<<dim3(NSEQ / 16, NHEADS, BATCH), blk, 0, stream>>>(
        Qb, Kb, Vt, Ob);
    out_proj_kernel<<<dim3(DIMC / 64, MROWS / 32), blk, 0, stream>>>(
        Ob, Wph, bp, (float*)d_out);
}
